// DctLSTM_85899345920341
// MI455X (gfx1250) — matrix operand order repaired and checked by reading, not compiled, not run on hardware
//
#include <hip/hip_runtime.h>

#define T_STEPS 512
#define BATCH   64
#define IN_DIM  1024
#define HID     1024
#define NNZ     523776
#define KDIM    2048          // IN_DIM + HID
#define NWG     128           // persistent workgroups
#define WG_THREADS 256        // 8 waves (wave32)
#define COLS_PER_WG 32        // 4 gates * 8 hidden units
#define HID_PER_WG  8

typedef __attribute__((ext_vector_type(16))) __bf16 v16bf;
typedef __attribute__((ext_vector_type(8)))  __bf16 bfx8;
typedef __attribute__((ext_vector_type(8)))  float  v8f;
typedef __attribute__((ext_vector_type(2)))  float  v2f;

// ---- LDS layout for the persistent LSTM kernel (bytes) ----
#define SMEM_W     0                       // 32 rows * 4096 B (2048 bf16) = 131072
#define SMEM_Z     131072                  // z tile: 64 x 33 floats (padded) = 8448
#define SMEM_C     (131072 + 8448)         // c state: 64 x 8 floats = 2048
#define SMEM_BIAS  (131072 + 8448 + 2048)  // 32 floats = 128
#define SMEM_BYTES (131072 + 8448 + 2048 + 128)

__device__ __forceinline__ float sigf(float x) { return 1.0f / (1.0f + __expf(-x)); }

__device__ __forceinline__ v16bf mk16(bfx8 lo, bfx8 hi) {
  union { v16bf v; bfx8 h[2]; } u;
  u.h[0] = lo; u.h[1] = hi;
  return u.v;
}

// ---------------------------------------------------------------------------
// Workspace init: zero Wf (scatter target), h ping-pong buffer 0, barrier ctr
// ---------------------------------------------------------------------------
__global__ void init_ws_kernel(float* __restrict__ Wf, unsigned int* __restrict__ h0z,
                               int* __restrict__ counter) {
  unsigned int stride = gridDim.x * blockDim.x;
  for (unsigned int i = blockIdx.x * blockDim.x + threadIdx.x; i < 4u * 1024u * 1024u; i += stride)
    Wf[i] = 0.0f;
  for (unsigned int i = blockIdx.x * blockDim.x + threadIdx.x; i < 32768u; i += stride)
    h0z[i] = 0u;   // 65536 bf16 zeros
  if (blockIdx.x == 0 && threadIdx.x == 0) *counter = 0;
}

// ---------------------------------------------------------------------------
// IDCT basis M[n][k] = c_k * cos(pi*(2n+1)*k / 2N), N = 1024 (f64 for accuracy)
// ---------------------------------------------------------------------------
__global__ void build_idct_kernel(float* __restrict__ Mm) {
  int id = blockIdx.x * blockDim.x + threadIdx.x;
  if (id >= IN_DIM * IN_DIM) return;
  int n = id >> 10, k = id & 1023;
  double ck = (k == 0) ? 0.03125 : 0.04419417382415922;  // sqrt(1/1024), sqrt(2/1024)
  Mm[id] = (float)(ck * cos(3.14159265358979323846 * (2.0 * n + 1.0) * (double)k / 2048.0));
}

// ---------------------------------------------------------------------------
// Scatter coeffs into dense Wf with column reversal:
// triu_indices(1024, 1, 1024): idx(r,c) = r*1023 - r(r-1)/2 + (c-r-1); write at col 1023-c
// ---------------------------------------------------------------------------
__global__ void scatter_kernel(const float* __restrict__ co, float* __restrict__ Wf) {
  int id = blockIdx.x * blockDim.x + threadIdx.x;
  if (id >= 1024 * 1024) return;
  int r = id >> 10, c = id & 1023;
  if (c < r + 1) return;
  int idx = r * 1023 - (r * (r - 1)) / 2 + (c - r - 1);
#pragma unroll
  for (int g = 0; g < 4; ++g)
    Wf[(size_t)g * 1048576 + (size_t)r * 1024 + (1023 - c)] = co[(size_t)g * NNZ + idx];
}

// ---------------------------------------------------------------------------
// Synthesis GEMM 1: tmp[g] = M (1024x1024) @ Wf[g] (1024x1024), f32 WMMA 16x16x4
// ---------------------------------------------------------------------------
__global__ __launch_bounds__(256) void synth_gemm1_kernel(const float* __restrict__ Mm,
                                                          const float* __restrict__ Wf,
                                                          float* __restrict__ tmp) {
  int wid = blockIdx.x * 8 + (threadIdx.x >> 5);  // 0..16383
  int g  = wid >> 12;
  int tt = wid & 4095;
  int mt = tt >> 6, nt = tt & 63;
  int lane = threadIdx.x & 31, half = lane >> 4, l15 = lane & 15;
  const float* Arow = Mm + (size_t)(mt * 16 + l15) * 1024;
  const float* Bg   = Wf + (size_t)g * 1048576;
  int ncol = nt * 16 + l15;
  v8f acc = {};
  for (int kt = 0; kt < 256; ++kt) {
    int k0 = kt * 4 + half * 2;
    v2f a = *(const v2f*)(Arow + k0);
    v2f b;
    b[0] = Bg[(size_t)k0 * 1024 + ncol];
    b[1] = Bg[(size_t)(k0 + 1) * 1024 + ncol];
    acc = __builtin_amdgcn_wmma_f32_16x16x4_f32(false, a, false, b, (short)0, acc, false, false);
  }
  float* Og = tmp + (size_t)g * 1048576;
#pragma unroll
  for (int r = 0; r < 8; ++r)
    Og[(size_t)(mt * 16 + r + half * 8) * 1024 + ncol] = acc[r];
}

// ---------------------------------------------------------------------------
// Synthesis GEMM 2 + pack: W[g][a][b] = sum_n tmp[g][a][n] * M[b][n]  (B = M^T)
// Writes bf16 into the WMMA-friendly packed layout:
//   packed_row = (a>>3)*32 + g*8 + (a&7),  k = colblock + b  (row stride 2048)
// ---------------------------------------------------------------------------
__global__ __launch_bounds__(256) void synth_gemm2_kernel(const float* __restrict__ tmp,
                                                          const float* __restrict__ Mm,
                                                          __bf16* __restrict__ Wp,
                                                          int colblock) {
  int wid = blockIdx.x * 8 + (threadIdx.x >> 5);
  int g  = wid >> 12;
  int tt = wid & 4095;
  int mt = tt >> 6, nt = tt & 63;
  int lane = threadIdx.x & 31, half = lane >> 4, l15 = lane & 15;
  const float* Arow = tmp + (size_t)g * 1048576 + (size_t)(mt * 16 + l15) * 1024;
  const float* Brow = Mm + (size_t)(nt * 16 + l15) * 1024;  // B[k][n] = M[n][k]
  v8f acc = {};
  for (int kt = 0; kt < 256; ++kt) {
    int k0 = kt * 4 + half * 2;
    v2f a = *(const v2f*)(Arow + k0);
    v2f b = *(const v2f*)(Brow + k0);
    acc = __builtin_amdgcn_wmma_f32_16x16x4_f32(false, a, false, b, (short)0, acc, false, false);
  }
  int bcol = nt * 16 + l15;
#pragma unroll
  for (int r = 0; r < 8; ++r) {
    int arow = mt * 16 + r + half * 8;
    int prow = (arow >> 3) * 32 + g * 8 + (arow & 7);
    Wp[(size_t)prow * KDIM + colblock + bcol] = (__bf16)acc[r];
  }
}

// ---------------------------------------------------------------------------
// Convert input f32 -> bf16 once (after synthesis; aliases synthesis scratch)
// ---------------------------------------------------------------------------
__global__ void cvt_x_kernel(const float* __restrict__ x, __bf16* __restrict__ xb) {
  size_t stride = (size_t)gridDim.x * blockDim.x;
  for (size_t i = (size_t)blockIdx.x * blockDim.x + threadIdx.x;
       i < (size_t)T_STEPS * BATCH * IN_DIM; i += stride)
    xb[i] = (__bf16)x[i];
}

// ---------------------------------------------------------------------------
// Persistent LSTM: 128 WGs, each owns 8 hidden units (32 gate-cols), W slice
// pinned in LDS for all 512 steps, bf16 WMMA 16x16x32, global barrier per step
// ---------------------------------------------------------------------------
__global__ __launch_bounds__(WG_THREADS, 1)
void lstm_persistent_kernel(const __bf16* __restrict__ xbf,
                            const __bf16* __restrict__ Wp,
                            const float* __restrict__ bias,
                            float* __restrict__ out,
                            __bf16* __restrict__ h0,
                            __bf16* __restrict__ h1,
                            int* __restrict__ counter) {
  extern __shared__ char smem[];
  float* z_s = (float*)(smem + SMEM_Z);
  float* c_s = (float*)(smem + SMEM_C);
  float* b_s = (float*)(smem + SMEM_BIAS);

  const int tid  = threadIdx.x;
  const int wg   = blockIdx.x;
  const int lane = tid & 31;
  const int wv   = tid >> 5;       // 0..7
  const int mt   = wv & 3;         // batch tile (64 rows / 16)
  const int nt   = wv >> 2;        // 0..1 (32 cols / 16)
  const int half = lane >> 4;
  const int l15  = lane & 15;
  const int arow = mt * 16 + l15;  // batch row
  const int ncol = nt * 16 + l15;  // local weight row / z column (g*8+jj)

  // Stage this WG's 32 weight rows (2048 bf16 each) into LDS with a 16B/row
  // rotation swizzle (bank-conflict-free B reads), via the CDNA5 async
  // global->LDS DMA path (ASYNCcnt). Dynamic LDS starts at offset 0 (no
  // static LDS in this kernel), so raw byte offsets address the same space
  // the compiler's ds ops use.
  {
    unsigned long long wsrc =
        (unsigned long long)(Wp + (size_t)wg * COLS_PER_WG * KDIM);
    for (int c = tid; c < 8192; c += WG_THREADS) {   // 8192 x 16B chunks
      int n  = c >> 8;
      int kb = (c & 255) * 16;
      unsigned long long gaddr = wsrc + (unsigned long long)(n * 4096 + kb);
      int laddr = SMEM_W + n * 4096 + ((kb + n * 16) & 4095);
      asm volatile("global_load_async_to_lds_b128 %0, %1, off"
                   :: "v"(laddr), "v"(gaddr) : "memory");
    }
    asm volatile("s_wait_asynccnt 0" ::: "memory");
  }
  for (int i = tid; i < BATCH * HID_PER_WG; i += WG_THREADS) c_s[i] = 0.0f;
  if (tid < COLS_PER_WG) b_s[tid] = bias[(tid >> 3) * HID + wg * HID_PER_WG + (tid & 7)];
  __syncthreads();

  const int boff = SMEM_W + ncol * 4096;

  for (int t = 0; t < T_STEPS; ++t) {
    const __bf16* xt   = xbf + (size_t)t * (BATCH * IN_DIM);
    const __bf16* hcur = (t & 1) ? h1 : h0;
    __bf16*       hnxt = (t & 1) ? h0 : h1;

    // Two independent accumulation chains to break back-to-back WMMA RAW
    v8f acc0 = {};
    v8f acc1 = {};
#pragma unroll 4
    for (int kt = 0; kt < 64; kt += 2) {
#pragma unroll
      for (int u = 0; u < 2; ++u) {
        int ku = kt + u;
        // A fragment (16x32 bf16): lane holds K chunks [klo..+7] and [klo+16..+23]
        int k = ku * 32 + half * 8;
        const __bf16* ap = (ku < 32) ? (xt + (size_t)arow * IN_DIM + k)
                                     : (hcur + (size_t)arow * HID + (k - IN_DIM));
        bfx8 alo = *(const bfx8*)ap;
        bfx8 ahi = *(const bfx8*)(ap + 16);
        v16bf a = mk16(alo, ahi);

        // B fragment (32x16 bf16): lane n holds K chunks [klo..+7] and [klo+16..+23] (swizzled)
        int kb = (ku * 32 + half * 8) * 2;
        bfx8 blo = *(const bfx8*)(smem + boff + ((kb      + ncol * 16) & 4095));
        bfx8 bhi = *(const bfx8*)(smem + boff + ((kb + 32 + ncol * 16) & 4095));
        v16bf b = mk16(blo, bhi);

        if (u == 0)
          acc0 = __builtin_amdgcn_wmma_f32_16x16x32_bf16(false, a, false, b, (short)0,
                                                         acc0, false, false);
        else
          acc1 = __builtin_amdgcn_wmma_f32_16x16x32_bf16(false, a, false, b, (short)0,
                                                         acc1, false, false);
      }
    }
    v8f acc = acc0 + acc1;

    // Spill z tile (batch x 32 gate-cols) to LDS for the gate combine
#pragma unroll
    for (int r = 0; r < 8; ++r)
      z_s[(mt * 16 + r + half * 8) * 33 + ncol] = acc[r];

    // Warm next step's x slab (global_prefetch_b8) while we wait
    if (t + 1 < T_STEPS) {
      const __bf16* xn = xbf + (size_t)(t + 1) * (BATCH * IN_DIM);
      __builtin_prefetch((const void*)(xn + (size_t)tid * 128), 0, 1);
      __builtin_prefetch((const void*)(xn + 32768 + (size_t)tid * 128), 0, 1);
    }
    __syncthreads();

    // Gate combine: each (b, jj) pair handled by exactly one thread
    for (int p = tid; p < BATCH * HID_PER_WG; p += WG_THREADS) {
      int b = p >> 3, jj = p & 7;
      float hg = z_s[b * 33 +      jj] + b_s[     jj];
      float ig = z_s[b * 33 +  8 + jj] + b_s[ 8 + jj];
      float fg = z_s[b * 33 + 16 + jj] + b_s[16 + jj];
      float og = z_s[b * 33 + 24 + jj] + b_s[24 + jj];
      float cn = sigf(ig) * tanhf(hg) + sigf(fg) * c_s[p];
      float o  = sigf(og) * tanhf(cn);
      c_s[p] = cn;
      int jg = wg * HID_PER_WG + jj;
      out[(size_t)t * (BATCH * HID) + (size_t)b * HID + jg] = o;
      hnxt[(size_t)b * HID + jg] = (__bf16)o;
      if (t == T_STEPS - 1) {
        out[(size_t)T_STEPS * (BATCH * HID) + (size_t)b * HID + jg] = o;                        // h_last
        out[(size_t)T_STEPS * (BATCH * HID) + (size_t)BATCH * HID + (size_t)b * HID + jg] = cn; // c_last
      }
    }

    // Device-wide barrier (release h writes, acquire before next step's reads)
    if (t != T_STEPS - 1) {
      __threadfence();
      __syncthreads();
      if (tid == 0) {
        __hip_atomic_fetch_add(counter, 1, __ATOMIC_RELEASE, __HIP_MEMORY_SCOPE_AGENT);
        int target = NWG * (t + 1);
        while (__hip_atomic_load(counter, __ATOMIC_ACQUIRE, __HIP_MEMORY_SCOPE_AGENT) < target)
          __builtin_amdgcn_s_sleep(1);
      }
      __syncthreads();
    }
  }
}

// ---------------------------------------------------------------------------
extern "C" void kernel_launch(void* const* d_in, const int* in_sizes, int n_in,
                              void* d_out, int out_size, void* d_ws, size_t ws_size,
                              hipStream_t stream) {
  const float* x    = (const float*)d_in[0];
  const float* cin  = (const float*)d_in[1];
  const float* chid = (const float*)d_in[2];
  const float* bias = (const float*)d_in[3];
  float* out = (float*)d_out;
  char* ws = (char*)d_ws;

  // Workspace layout (xbf deliberately aliases synthesis scratch; stream-ordered):
  float*  Mm  = (float*)(ws);                    // [0, 4MB)   IDCT basis
  float*  Wf  = (float*)(ws + (4u << 20));       // [4, 20MB)  dense coeff matrices
  float*  tmp = (float*)(ws + (20u << 20));      // [20, 36MB) M @ Wf
  __bf16* xbf = (__bf16*)(ws);                   // [0, 64MB)  bf16 input (after synthesis)
  __bf16* Wp  = (__bf16*)(ws + (64u << 20));     // [64, 80MB) packed bf16 weights
  __bf16* h0  = (__bf16*)(ws + (80u << 20));     // h ping-pong (2 x 128KB)
  __bf16* h1  = h0 + BATCH * HID;
  int* counter = (int*)(ws + (80u << 20) + 262144);

  init_ws_kernel<<<2048, 256, 0, stream>>>(Wf, (unsigned int*)h0, counter);
  build_idct_kernel<<<4096, 256, 0, stream>>>(Mm);

  // w_ih -> packed K columns [0, 1024)
  scatter_kernel<<<4096, 256, 0, stream>>>(cin, Wf);
  synth_gemm1_kernel<<<2048, 256, 0, stream>>>(Mm, Wf, tmp);
  synth_gemm2_kernel<<<2048, 256, 0, stream>>>(tmp, Mm, Wp, 0);

  // w_hh -> packed K columns [1024, 2048)
  scatter_kernel<<<4096, 256, 0, stream>>>(chid, Wf);
  synth_gemm1_kernel<<<2048, 256, 0, stream>>>(Mm, Wf, tmp);
  synth_gemm2_kernel<<<2048, 256, 0, stream>>>(tmp, Mm, Wp, 1024);

  cvt_x_kernel<<<8192, 256, 0, stream>>>(x, xbf);

  (void)hipFuncSetAttribute((const void*)lstm_persistent_kernel,
                            hipFuncAttributeMaxDynamicSharedMemorySize, SMEM_BYTES);
  lstm_persistent_kernel<<<NWG, WG_THREADS, SMEM_BYTES, stream>>>(xbf, Wp, bias, out,
                                                                  h0, h1, counter);
}
